// TranslatedKeySelfAttention_69827578298378
// MI455X (gfx1250) — matrix operand order repaired and checked by reading, not compiled, not run on hardware
//
#include <hip/hip_runtime.h>

#define HIDDEN 768
#define NHEADS 12
#define HDIM   64
#define BATCH  4
#define SEQ    2048
#define DEGREE 16
#define MROWS  (BATCH * SEQ)      /* 8192 */
#define EDGES  (MROWS * DEGREE)   /* 131072 */

typedef __attribute__((ext_vector_type(16))) __bf16 v16bf;
typedef __attribute__((ext_vector_type(8)))  __bf16 v8bf;
typedef __attribute__((ext_vector_type(4)))  __bf16 v4bf;
typedef __attribute__((ext_vector_type(8)))  float  v8f;

// float -> bf16, round-to-nearest-even
__device__ __forceinline__ __bf16 f2bf(float x) {
    unsigned u = __builtin_bit_cast(unsigned, x);
    unsigned r = u + 0x7FFFu + ((u >> 16) & 1u);
    unsigned short h = (unsigned short)(r >> 16);
    return __builtin_bit_cast(__bf16, h);
}
__device__ __forceinline__ float bf2f(__bf16 b) {
    unsigned short h = __builtin_bit_cast(unsigned short, b);
    unsigned u = ((unsigned)h) << 16;
    return __builtin_bit_cast(float, u);
}

// ---------------------------------------------------------------------------
// Pre-pass: split an f32 array into bf16 hi/lo planes (x = hi + lo).
// Memory-bound, runs once; removes all conversion VALU from the GEMM loop.
// ---------------------------------------------------------------------------
__global__ __launch_bounds__(256)
void split_bf16_kernel(const float* __restrict__ src,
                       __bf16* __restrict__ hi, __bf16* __restrict__ lo,
                       int n4) {
    const int i = blockIdx.x * 256 + threadIdx.x;
    if (i >= n4) return;
    const float4 v = ((const float4*)src)[i];
    float f[4] = {v.x, v.y, v.z, v.w};
    v4bf h, l;
#pragma unroll
    for (int j = 0; j < 4; ++j) {
        __bf16 hh = f2bf(f[j]);
        h[j] = hh;
        l[j] = f2bf(f[j] - bf2f(hh));
    }
    ((v4bf*)hi)[i] = h;
    ((v4bf*)lo)[i] = l;
}

// ---------------------------------------------------------------------------
// Kernel 1 (fast path): Q/K/V projection GEMM from pre-split bf16 planes.
// y = x @ W.T + b via split-bf16: acc += Ahi*Bhi + Alo*Bhi + Ahi*Blo.
// Block = 256 threads = 8 wave32s; each wave owns a 16x64 output strip.
// grid = (MROWS/128, HIDDEN/64, 3); blockIdx.z selects {q,k,v}.
// ---------------------------------------------------------------------------
__global__ __launch_bounds__(256)
void qkv_gemm_bf16_kernel(const __bf16* __restrict__ Xh, const __bf16* __restrict__ Xl,
                          const __bf16* __restrict__ Wh3, const __bf16* __restrict__ Wl3,
                          const float* __restrict__ bq, const float* __restrict__ bk,
                          const float* __restrict__ bv,
                          float* __restrict__ out_base) {
    const int lane    = threadIdx.x & 31;
    const int wave    = threadIdx.x >> 5;
    const int rowBase = blockIdx.x * 128 + wave * 16;
    const int colBase = blockIdx.y * 64;
    const int which   = blockIdx.z;

    const float* bias = (which == 0) ? bq : (which == 1) ? bk : bv;
    float* out = out_base + (size_t)which * MROWS * HIDDEN;
    const __bf16* Whb = Wh3 + (size_t)which * HIDDEN * HIDDEN;
    const __bf16* Wlb = Wl3 + (size_t)which * HIDDEN * HIDDEN;

    const int halfSel = lane >> 4;   // 0: lanes 0-15, 1: lanes 16-31
    const int lane15  = lane & 15;

    // A layout (16-bit A 16x32): lanes 0-15 hold K = k0+0..15 of
    // row M = lane; lanes 16-31 hold K = k0+16..31 (contiguous in k).
    const __bf16* XhRow = Xh + (size_t)(rowBase + lane15) * HIDDEN + halfSel * 16;
    const __bf16* XlRow = Xl + (size_t)(rowBase + lane15) * HIDDEN + halfSel * 16;

    v8f acc[4] = {};

    for (int k0 = 0; k0 < HIDDEN; k0 += 32) {
        const v8bf ah0 = *(const v8bf*)(XhRow + k0);
        const v8bf ah1 = *(const v8bf*)(XhRow + k0 + 8);
        const v8bf al0 = *(const v8bf*)(XlRow + k0);
        const v8bf al1 = *(const v8bf*)(XlRow + k0 + 8);
        const v16bf a_hi = __builtin_shufflevector(ah0, ah1,
            0,1,2,3,4,5,6,7,8,9,10,11,12,13,14,15);
        const v16bf a_lo = __builtin_shufflevector(al0, al1,
            0,1,2,3,4,5,6,7,8,9,10,11,12,13,14,15);

#pragma unroll
        for (int j = 0; j < 4; ++j) {
            // B = W.T: (k, n) = W[n][k]; lanes 0-15: N = lane, K = k0..k0+15;
            // lanes 16-31: N = lane-16, K = k0+16..k0+31 (contiguous in k).
            const size_t woff = (size_t)(colBase + j * 16 + lane15) * HIDDEN
                                + k0 + halfSel * 16;
            const v8bf bh0 = *(const v8bf*)(Whb + woff);
            const v8bf bh1 = *(const v8bf*)(Whb + woff + 8);
            const v8bf bl0 = *(const v8bf*)(Wlb + woff);
            const v8bf bl1 = *(const v8bf*)(Wlb + woff + 8);
            const v16bf b_hi = __builtin_shufflevector(bh0, bh1,
                0,1,2,3,4,5,6,7,8,9,10,11,12,13,14,15);
            const v16bf b_lo = __builtin_shufflevector(bl0, bl1,
                0,1,2,3,4,5,6,7,8,9,10,11,12,13,14,15);

            acc[j] = __builtin_amdgcn_wmma_f32_16x16x32_bf16(
                         false, a_hi, false, b_hi, (short)0, acc[j], false, false);
            acc[j] = __builtin_amdgcn_wmma_f32_16x16x32_bf16(
                         false, a_lo, false, b_hi, (short)0, acc[j], false, false);
            acc[j] = __builtin_amdgcn_wmma_f32_16x16x32_bf16(
                         false, a_hi, false, b_lo, (short)0, acc[j], false, false);
        }
    }

    // C/D layout: VGPR r -> M = r + 8*halfSel, N = lane15
#pragma unroll
    for (int j = 0; j < 4; ++j) {
        const int n = colBase + j * 16 + lane15;
        const float bval = bias[n];
#pragma unroll
        for (int r = 0; r < 8; ++r) {
            const int m = rowBase + r + halfSel * 8;
            out[(size_t)m * HIDDEN + n] = acc[j][r] + bval;
        }
    }
}

// ---------------------------------------------------------------------------
// Kernel 1 (fallback, small-ws): same GEMM with inline f32->bf16 splitting.
// ---------------------------------------------------------------------------
__global__ __launch_bounds__(256)
void qkv_gemm_kernel(const float* __restrict__ X,
                     const float* __restrict__ Wq, const float* __restrict__ bq,
                     const float* __restrict__ Wk, const float* __restrict__ bk,
                     const float* __restrict__ Wv, const float* __restrict__ bv,
                     float* __restrict__ out_base) {
    const int lane    = threadIdx.x & 31;
    const int wave    = threadIdx.x >> 5;
    const int rowBase = blockIdx.x * 128 + wave * 16;
    const int colBase = blockIdx.y * 64;
    const int which   = blockIdx.z;

    const float* W    = (which == 0) ? Wq : (which == 1) ? Wk : Wv;
    const float* bias = (which == 0) ? bq : (which == 1) ? bk : bv;
    float* out = out_base + (size_t)which * MROWS * HIDDEN;

    const int halfSel = lane >> 4;
    const int lane15  = lane & 15;
    const float* Xrow = X + (size_t)(rowBase + lane15) * HIDDEN + halfSel * 16;

    v8f acc[4] = {};
    for (int k0 = 0; k0 < HIDDEN; k0 += 32) {
        float af[16];
#pragma unroll
        for (int i = 0; i < 8; ++i) af[i]     = Xrow[k0 + i];
#pragma unroll
        for (int i = 0; i < 8; ++i) af[8 + i] = Xrow[k0 + 8 + i];
        v16bf a_hi, a_lo;
#pragma unroll
        for (int i = 0; i < 16; ++i) {
            __bf16 h = f2bf(af[i]);
            a_hi[i] = h;
            a_lo[i] = f2bf(af[i] - bf2f(h));
        }
#pragma unroll
        for (int j = 0; j < 4; ++j) {
            const int bcol = colBase + j * 16 + lane15;
            const float* Wrow = W + (size_t)bcol * HIDDEN + k0 + halfSel * 16;
            float bf[16];
#pragma unroll
            for (int i = 0; i < 16; ++i) bf[i] = Wrow[i];
            v16bf b_hi, b_lo;
#pragma unroll
            for (int i = 0; i < 16; ++i) {
                __bf16 h = f2bf(bf[i]);
                b_hi[i] = h;
                b_lo[i] = f2bf(bf[i] - bf2f(h));
            }
            acc[j] = __builtin_amdgcn_wmma_f32_16x16x32_bf16(
                         false, a_hi, false, b_hi, (short)0, acc[j], false, false);
            acc[j] = __builtin_amdgcn_wmma_f32_16x16x32_bf16(
                         false, a_lo, false, b_hi, (short)0, acc[j], false, false);
            acc[j] = __builtin_amdgcn_wmma_f32_16x16x32_bf16(
                         false, a_hi, false, b_lo, (short)0, acc[j], false, false);
        }
    }
#pragma unroll
    for (int j = 0; j < 4; ++j) {
        const int n = colBase + j * 16 + lane15;
        const float bval = bias[n];
#pragma unroll
        for (int r = 0; r < 8; ++r) {
            const int m = rowBase + r + halfSel * 8;
            out[(size_t)m * HIDDEN + n] = acc[j][r] + bval;
        }
    }
}

// ---------------------------------------------------------------------------
// Kernel 2: translated-key edge attention. One wave32 per (b, i) node.
// ---------------------------------------------------------------------------
__global__ __launch_bounds__(256)
void edge_attn_kernel(const float* __restrict__ Q, const float* __restrict__ K,
                      const float* __restrict__ V, const int* __restrict__ eidx,
                      const float* __restrict__ rel, float* __restrict__ out) {
    __shared__ float qs[8][HIDDEN];
    __shared__ float pr[8][DEGREE * NHEADS];

    const int lane  = threadIdx.x & 31;
    const int wave  = threadIdx.x >> 5;
    const int node  = blockIdx.x * 8 + wave;     // node = b*SEQ + i
    const int ebase = node * DEGREE;
    const int bb    = node / SEQ;

    for (int c = lane; c < HIDDEN; c += 32)
        qs[wave][c] = Q[(size_t)node * HIDDEN + c];

    int te[DEGREE], re[DEGREE];
#pragma unroll
    for (int e = 0; e < DEGREE; ++e) {
        te[e] = eidx[2 * EDGES + ebase + e];
        re[e] = eidx[3 * EDGES + ebase + e];
    }
    __syncthreads();

    for (int h = 0; h < NHEADS; ++h) {
        const int d0 = h * HDIM + 2 * lane;
        const float q0 = qs[wave][d0];
        const float q1 = qs[wave][d0 + 1];
#pragma unroll
        for (int e = 0; e < DEGREE; ++e) {
            const size_t krow = ((size_t)bb * SEQ + te[e]) * HIDDEN;
            const size_t rrow = (size_t)re[e] * HIDDEN;
            float p = q0 * (K[krow + d0]     - rel[rrow + d0])
                    + q1 * (K[krow + d0 + 1] - rel[rrow + d0 + 1]);
#pragma unroll
            for (int off = 16; off >= 1; off >>= 1)
                p += __shfl_xor(p, off, 32);
            if (lane == 0)
                pr[wave][e * NHEADS + h] = p * 0.125f;   // / sqrt(64)
        }
    }
    __syncthreads();

    if (lane < NHEADS) {
        float m = -3.0e38f;
#pragma unroll
        for (int e = 0; e < DEGREE; ++e)
            m = fmaxf(m, pr[wave][e * NHEADS + lane]);
        float ex[DEGREE];
        float s = 0.f;
#pragma unroll
        for (int e = 0; e < DEGREE; ++e) {
            ex[e] = __expf(pr[wave][e * NHEADS + lane] - m);
            s += ex[e];
        }
        const float inv = 1.0f / s;
#pragma unroll
        for (int e = 0; e < DEGREE; ++e)
            pr[wave][e * NHEADS + lane] = ex[e] * inv;
    }
    __syncthreads();

    for (int c = lane; c < HIDDEN; c += 32) {
        const int h = c >> 6;
        float acc = 0.f;
#pragma unroll
        for (int e = 0; e < DEGREE; ++e)
            acc += pr[wave][e * NHEADS + h] *
                   V[((size_t)bb * SEQ + te[e]) * HIDDEN + c];
        out[(size_t)node * HIDDEN + c] = acc;
    }
}

// ---------------------------------------------------------------------------
extern "C" void kernel_launch(void* const* d_in, const int* in_sizes, int n_in,
                              void* d_out, int out_size, void* d_ws, size_t ws_size,
                              hipStream_t stream) {
    const float* X    = (const float*)d_in[0];
    const int*   eidx = (const int*)  d_in[1];
    const float* Wq   = (const float*)d_in[2];
    const float* bq   = (const float*)d_in[3];
    const float* Wk   = (const float*)d_in[4];
    const float* bk   = (const float*)d_in[5];
    const float* Wv   = (const float*)d_in[6];
    const float* bv   = (const float*)d_in[7];
    const float* rel  = (const float*)d_in[8];

    const size_t nX = (size_t)MROWS * HIDDEN;       // 6,291,456
    const size_t nW = (size_t)HIDDEN * HIDDEN;      //   589,824

    float* ws = (float*)d_ws;
    float* Qb = ws;
    float* Kb = ws + nX;
    float* Vb = ws + 2 * nX;

    // bf16 planes live after the f32 Q/K/V region
    __bf16* planes = (__bf16*)(ws + 3 * nX);
    __bf16* Xh  = planes;
    __bf16* Xl  = Xh + nX;
    __bf16* Wh3 = Xl + nX;          // 3 matrices concatenated (q,k,v)
    __bf16* Wl3 = Wh3 + 3 * nW;

    const size_t need = 3 * nX * sizeof(float)
                      + (2 * nX + 6 * nW) * sizeof(__bf16);

    dim3 g1(MROWS / 128, HIDDEN / 64, 3);    // 64 x 12 x 3 blocks

    if (ws_size >= need) {
        // pre-split X and the three W matrices into bf16 hi/lo planes
        const int nX4 = (int)(nX / 4);
        const int nW4 = (int)(nW / 4);
        split_bf16_kernel<<<(nX4 + 255) / 256, 256, 0, stream>>>(X,  Xh,            Xl,            nX4);
        split_bf16_kernel<<<(nW4 + 255) / 256, 256, 0, stream>>>(Wq, Wh3,           Wl3,           nW4);
        split_bf16_kernel<<<(nW4 + 255) / 256, 256, 0, stream>>>(Wk, Wh3 + nW,      Wl3 + nW,      nW4);
        split_bf16_kernel<<<(nW4 + 255) / 256, 256, 0, stream>>>(Wv, Wh3 + 2 * nW,  Wl3 + 2 * nW,  nW4);
        qkv_gemm_bf16_kernel<<<g1, 256, 0, stream>>>(Xh, Xl, Wh3, Wl3, bq, bk, bv, ws);
    } else {
        // fallback: inline conversion (higher VALU, still correct)
        qkv_gemm_kernel<<<g1, 256, 0, stream>>>(X, Wq, bq, Wk, bk, Wv, bv, ws);
    }

    dim3 g2(MROWS / 8);                      // 1024 blocks, 8 waves each
    edge_attn_kernel<<<g2, 256, 0, stream>>>(Qb, Kb, Vb, eidx, rel, (float*)d_out);
}
